// SRNNGLN_Cell_69252052680710
// MI455X (gfx1250) — matrix operand order repaired and checked by reading, not compiled, not run on hardware
//
#include <hip/hip_runtime.h>
#include <math.h>

typedef __bf16 v16bf __attribute__((ext_vector_type(16)));
typedef __bf16 v8bf  __attribute__((ext_vector_type(8)));
typedef __bf16 v4bf  __attribute__((ext_vector_type(4)));
typedef float  v8f   __attribute__((ext_vector_type(8)));

#define BB   16
#define SS   4096
#define II   256
#define HY   64
#define HH   512
#define NROWS (BB*SS)   // 65536

// ---- CDNA5 async global->LDS path (guarded; falls back to plain copies) ----
#if defined(__gfx1250__) && \
    __has_builtin(__builtin_amdgcn_global_load_async_to_lds_b128) && \
    __has_builtin(__builtin_amdgcn_s_wait_asynccnt)
#define ASYNC_LDS 1
#endif

typedef int v4i_ __attribute__((vector_size(4 * sizeof(int))));
typedef __attribute__((address_space(1))) v4i_ gv4i;   // global
typedef __attribute__((address_space(3))) v4i_ sv4i;   // LDS

__device__ __forceinline__ void lds_fill_16B(void* dst_lds, const void* src_glb) {
#ifdef ASYNC_LDS
    gv4i* g = (gv4i*)(unsigned long long)(const char*)src_glb;
    sv4i* l = (sv4i*)(unsigned int)(unsigned long long)(char*)dst_lds;
    __builtin_amdgcn_global_load_async_to_lds_b128(g, l, 0, 0);
#else
    *(int4*)dst_lds = *(const int4*)src_glb;
#endif
}
__device__ __forceinline__ void lds_fill_wait() {
#ifdef ASYNC_LDS
    __builtin_amdgcn_s_wait_asynccnt(0);
#endif
    __syncthreads();
}

__device__ __forceinline__ float sigmoidf_(float x) {
    return 1.0f / (1.0f + __expf(-x));
}

__device__ __forceinline__ v8f wmma_bf16(v16bf a, v16bf b, v8f c) {
    // D = A(16x32 bf16) x B(32x16 bf16) + C(16x16 f32)
    return __builtin_amdgcn_wmma_f32_16x16x32_bf16(
        false, a, false, b, (short)0, c, false, false);
}

// A fragment (16x32) from row-major bf16 matrix, row stride ldk elements.
__device__ __forceinline__ v16bf load_a_frag_bf16(const __bf16* __restrict__ base, int ldk,
                                                  int row, int k0, int laneHi) {
    const __bf16* p = base + (size_t)row * ldk + k0 + laneHi * 8;
    v8bf lo = *(const v8bf*)(p);
    v8bf hi = *(const v8bf*)(p + 16);
    v16bf a;
#pragma unroll
    for (int i = 0; i < 8; ++i) { a[i] = lo[i]; a[8 + i] = hi[i]; }
    return a;
}

// B fragment (32x16) from row-major bf16 weight W[ncol][k] (LDS resident).
// lane col n = n0+laneLo ; elem j<8 : K = k0 + laneHi*8 + j ; j>=8 : K = k0 + 16 + laneHi*8 + (j-8)
__device__ __forceinline__ v16bf load_b_frag_lds(const __bf16* W, int ldk,
                                                 int n0, int k0, int laneLo, int laneHi) {
    const __bf16* p = W + (n0 + laneLo) * ldk + k0 + laneHi * 8;
    v8bf lo = *(const v8bf*)(p);
    v8bf hi = *(const v8bf*)(p + 16);
    v16bf b;
#pragma unroll
    for (int i = 0; i < 8; ++i) { b[i] = lo[i]; b[8 + i] = hi[i]; }
    return b;
}

// raw f32 A-tile gather (4x float4) for software pipelining
struct AFrag32 { float4 f0, f1, f2, f3; };
__device__ __forceinline__ AFrag32 load_a_raw(const float* __restrict__ x, int row, int k0, int laneHi) {
    const float* p = x + (size_t)row * II + k0 + laneHi * 8;
    AFrag32 r;
    r.f0 = *(const float4*)(p);
    r.f1 = *(const float4*)(p + 4);
    r.f2 = *(const float4*)(p + 16);
    r.f3 = *(const float4*)(p + 20);
    return r;
}

// ---------------------------------------------------------------------------
// Kernel 0: streaming f32 -> bf16 conversion (weights pre-pass)
// ---------------------------------------------------------------------------
__global__ __launch_bounds__(256) void cvt_bf16_kernel(const float* __restrict__ src,
                                                       __bf16* __restrict__ dst, int n4) {
    int i = (blockIdx.x * 256 + threadIdx.x);
    if (i < n4) {
        float4 f = ((const float4*)src)[i];
        v4bf o;
        o[0] = (__bf16)f.x; o[1] = (__bf16)f.y; o[2] = (__bf16)f.z; o[3] = (__bf16)f.w;
        ((v4bf*)dst)[i] = o;
    }
}

// ---------------------------------------------------------------------------
// Kernel 1: g[n,h] = relu( (x@lin0_W^T + b) * sigmoid(x@sig0_W^T + b) )
// Weights staged in LDS (64 KB, async); A fragment software-pipelined; writes
// the bf16 copy xb of x as a byproduct for kernel 2.
// ---------------------------------------------------------------------------
__global__ __launch_bounds__(256) void gln_g_kernel(
    const float* __restrict__ x,
    const __bf16* __restrict__ wl,   // lin0_W bf16 [64][256]
    const __bf16* __restrict__ ws,   // sig0_W bf16 [64][256]
    const float* __restrict__ lin0_b, const float* __restrict__ sig0_b,
    __bf16* __restrict__ g,
    __bf16* __restrict__ xb)
{
    __shared__ __align__(16) __bf16 s_wl[HY * II];  // 32 KB
    __shared__ __align__(16) __bf16 s_ws[HY * II];  // 32 KB

    const int tid = threadIdx.x;
    {
        const int n16 = HY * II * 2 / 16;   // 2048 x 16B per matrix
        int4* dl = (int4*)s_wl;  const int4* sl = (const int4*)wl;
        int4* dsm = (int4*)s_ws; const int4* sm = (const int4*)ws;
        for (int i = tid; i < n16; i += 256) {
            lds_fill_16B(dl + i, sl + i);
            lds_fill_16B(dsm + i, sm + i);
        }
        lds_fill_wait();
    }

    const int lane = tid & 31;
    const int laneLo = lane & 15;
    const int laneHi = lane >> 4;
    const int wave = tid >> 5;
    const int mBase = (blockIdx.x * 8 + wave) * 16;
    const int row = mBase + laneLo;

    v8f accL[4] = {};
    v8f accS[4] = {};

    AFrag32 cur = load_a_raw(x, row, 0, laneHi);
    for (int k0 = 0; k0 < II; k0 += 32) {
        AFrag32 nxt = (k0 + 32 < II) ? load_a_raw(x, row, k0 + 32, laneHi) : cur;

        v8bf lo, hi;
        lo[0]=(__bf16)cur.f0.x; lo[1]=(__bf16)cur.f0.y; lo[2]=(__bf16)cur.f0.z; lo[3]=(__bf16)cur.f0.w;
        lo[4]=(__bf16)cur.f1.x; lo[5]=(__bf16)cur.f1.y; lo[6]=(__bf16)cur.f1.z; lo[7]=(__bf16)cur.f1.w;
        hi[0]=(__bf16)cur.f2.x; hi[1]=(__bf16)cur.f2.y; hi[2]=(__bf16)cur.f2.z; hi[3]=(__bf16)cur.f2.w;
        hi[4]=(__bf16)cur.f3.x; hi[5]=(__bf16)cur.f3.y; hi[6]=(__bf16)cur.f3.z; hi[7]=(__bf16)cur.f3.w;
        __bf16* q = xb + (size_t)row * II + k0 + laneHi * 8;
        *(v8bf*)(q) = lo;
        *(v8bf*)(q + 16) = hi;
        v16bf a;
#pragma unroll
        for (int i = 0; i < 8; ++i) { a[i] = lo[i]; a[8 + i] = hi[i]; }

#pragma unroll
        for (int c = 0; c < 4; ++c) {
            v16bf bl = load_b_frag_lds(s_wl, II, c * 16, k0, laneLo, laneHi);
            accL[c] = wmma_bf16(a, bl, accL[c]);
            v16bf bs = load_b_frag_lds(s_ws, II, c * 16, k0, laneLo, laneHi);
            accS[c] = wmma_bf16(a, bs, accS[c]);
        }
        cur = nxt;
    }

#pragma unroll
    for (int c = 0; c < 4; ++c) {
        const int h = c * 16 + laneLo;
        const float bl = lin0_b[h];
        const float bs = sig0_b[h];
#pragma unroll
        for (int r = 0; r < 8; ++r) {
            const int m = mBase + r + 8 * laneHi;
            float lin = accL[c][r] + bl;
            float gate = sigmoidf_(accS[c][r] + bs);
            float gv = lin * gate;
            gv = gv > 0.0f ? gv : 0.0f;
            g[(size_t)m * HY + h] = (__bf16)gv;
        }
    }
}

// ---------------------------------------------------------------------------
// Kernel 2: bmat[n,d] = (g@fch_W^T + fch_b) * sigmoid(x@fc2_W^T + fc2_b)
// d-slice of weights staged in LDS (40 KB, async); A fragments pipelined.
// ---------------------------------------------------------------------------
__global__ __launch_bounds__(256) void bpre_kernel(
    const __bf16* __restrict__ xb,
    const __bf16* __restrict__ g,
    const __bf16* __restrict__ wfch,  // bf16 [512][64]
    const __bf16* __restrict__ wfc2,  // bf16 [512][256]
    const float* __restrict__ fch_b, const float* __restrict__ fc2_b,
    float* __restrict__ bmat)
{
    __shared__ __align__(16) __bf16 s_wh[64 * HY];  //  8 KB  (rows d0..d0+63)
    __shared__ __align__(16) __bf16 s_w2[64 * II];  // 32 KB

    const int tid = threadIdx.x;
    const int d0 = blockIdx.y * 64;
    {
        const int4* sh = (const int4*)(wfch + (size_t)d0 * HY);
        const int4* s2 = (const int4*)(wfc2 + (size_t)d0 * II);
        int4* dh = (int4*)s_wh;
        int4* d2 = (int4*)s_w2;
        const int nh = 64 * HY * 2 / 16;   // 512
        const int n2 = 64 * II * 2 / 16;   // 2048
        for (int i = tid; i < nh; i += 256) lds_fill_16B(dh + i, sh + i);
        for (int i = tid; i < n2; i += 256) lds_fill_16B(d2 + i, s2 + i);
        lds_fill_wait();
    }

    const int lane = tid & 31;
    const int laneLo = lane & 15;
    const int laneHi = lane >> 4;
    const int wave = tid >> 5;
    const int mBase = (blockIdx.x * 8 + wave) * 16;
    const int row = mBase + laneLo;

    v8f accB[4] = {};
    v8f accS[4] = {};

    // g @ fch_W^T  (K = 64)
    {
        v16bf a = load_a_frag_bf16(g, HY, row, 0, laneHi);
        for (int k0 = 0; k0 < HY; k0 += 32) {
            v16bf an = (k0 + 32 < HY) ? load_a_frag_bf16(g, HY, row, k0 + 32, laneHi) : a;
#pragma unroll
            for (int c = 0; c < 4; ++c) {
                v16bf bw = load_b_frag_lds(s_wh, HY, c * 16, k0, laneLo, laneHi);
                accB[c] = wmma_bf16(a, bw, accB[c]);
            }
            a = an;
        }
    }
    // x @ fc2_W^T  (K = 256)
    {
        v16bf a = load_a_frag_bf16(xb, II, row, 0, laneHi);
        for (int k0 = 0; k0 < II; k0 += 32) {
            v16bf an = (k0 + 32 < II) ? load_a_frag_bf16(xb, II, row, k0 + 32, laneHi) : a;
#pragma unroll
            for (int c = 0; c < 4; ++c) {
                v16bf bw = load_b_frag_lds(s_w2, II, c * 16, k0, laneLo, laneHi);
                accS[c] = wmma_bf16(a, bw, accS[c]);
            }
            a = an;
        }
    }

#pragma unroll
    for (int c = 0; c < 4; ++c) {
        const int d = d0 + c * 16 + laneLo;
        const float bb = fch_b[d];
        const float sb = fc2_b[d];
#pragma unroll
        for (int r = 0; r < 8; ++r) {
            const int m = mBase + r + 8 * laneHi;
            float val = (accB[c][r] + bb) * sigmoidf_(accS[c][r] + sb);
            bmat[(size_t)m * HH + d] = val;
        }
    }
}

// ---------------------------------------------------------------------------
// Kernel 3: h_t = relu(b_t + roll(h_{t-1}, 1));  outputs + last.
// ---------------------------------------------------------------------------
__global__ __launch_bounds__(512) void recur_kernel(
    const float* __restrict__ bmat,
    const float* __restrict__ hidden,
    float* __restrict__ out,
    float* __restrict__ last)
{
    __shared__ float hbuf[2][HH];
    const int b = blockIdx.x;
    const int j = threadIdx.x;

    hbuf[0][j] = hidden[(size_t)b * HH + j];
    __syncthreads();

    const float* bp = bmat + (size_t)b * SS * HH + j;
    float* op = out + (size_t)b * SS * HH + j;

    int p = 0;
    float bv = bp[0];
    float hn = 0.0f;
    for (int t = 0; t < SS; ++t) {
        float bnext = (t + 1 < SS) ? bp[(size_t)(t + 1) * HH] : 0.0f;
        if (t + 8 < SS) __builtin_prefetch(bp + (size_t)(t + 8) * HH, 0, 0);
        float hprev = hbuf[p][(j + HH - 1) & (HH - 1)];
        hn = fmaxf(bv + hprev, 0.0f);
        op[(size_t)t * HH] = hn;
        hbuf[p ^ 1][j] = hn;
        __syncthreads();
        p ^= 1;
        bv = bnext;
    }
    last[(size_t)b * HH + j] = hn;
}

// ---------------------------------------------------------------------------
extern "C" void kernel_launch(void* const* d_in, const int* in_sizes, int n_in,
                              void* d_out, int out_size, void* d_ws, size_t ws_size,
                              hipStream_t stream) {
    const float* x      = (const float*)d_in[0];
    const float* hidden = (const float*)d_in[1];
    const float* lin0_W = (const float*)d_in[2];
    const float* lin0_b = (const float*)d_in[3];
    const float* sig0_W = (const float*)d_in[4];
    const float* sig0_b = (const float*)d_in[5];
    const float* fch_W  = (const float*)d_in[6];
    const float* fch_b  = (const float*)d_in[7];
    const float* fc2_W  = (const float*)d_in[8];
    const float* fc2_b  = (const float*)d_in[9];

    char* wsb = (char*)d_ws;
    __bf16* xb  = (__bf16*)wsb;  wsb += (size_t)NROWS * II * 2;
    __bf16* g   = (__bf16*)wsb;  wsb += (size_t)NROWS * HY * 2;
    __bf16* w_l = (__bf16*)wsb;  wsb += (size_t)HY * II * 2;
    __bf16* w_s = (__bf16*)wsb;  wsb += (size_t)HY * II * 2;
    __bf16* w_h = (__bf16*)wsb;  wsb += (size_t)HH * HY * 2;
    __bf16* w_2 = (__bf16*)wsb;  wsb += (size_t)HH * II * 2;
    float*  bmat = (float*)wsb;

    float* outp = (float*)d_out;
    float* lastp = outp + (size_t)NROWS * HH;

    dim3 blk(256);
    cvt_bf16_kernel<<<dim3((HY * II / 4 + 255) / 256), blk, 0, stream>>>(lin0_W, w_l, HY * II / 4);
    cvt_bf16_kernel<<<dim3((HY * II / 4 + 255) / 256), blk, 0, stream>>>(sig0_W, w_s, HY * II / 4);
    cvt_bf16_kernel<<<dim3((HH * HY / 4 + 255) / 256), blk, 0, stream>>>(fch_W, w_h, HH * HY / 4);
    cvt_bf16_kernel<<<dim3((HH * II / 4 + 255) / 256), blk, 0, stream>>>(fc2_W, w_2, HH * II / 4);

    dim3 grdA(NROWS / (16 * 8), 1);       // 512 blocks, wave -> 16x64 tile
    gln_g_kernel<<<grdA, blk, 0, stream>>>(x, w_l, w_s, lin0_b, sig0_b, g, xb);

    dim3 grdB(NROWS / (16 * 8), HH / 64); // 512 x 8
    bpre_kernel<<<grdB, blk, 0, stream>>>(xb, g, w_h, w_2, fch_b, fc2_b, bmat);

    recur_kernel<<<dim3(BB), dim3(HH), 0, stream>>>(bmat, hidden, outp, lastp);
}
